// WindowAttention_66468913873024
// MI455X (gfx1250) — matrix operand order repaired and checked by reading, not compiled, not run on hardware
//
#include <hip/hip_runtime.h>

// ---------------------------------------------------------------------------
// SwinV2 window attention for gfx1250 (MI455X), wave32 + WMMA f16.
// B=4, H=W=256, C=144, heads=6, hd=24, ws=8, shift=4, N=64, windows=4096.
// ---------------------------------------------------------------------------

typedef __attribute__((ext_vector_type(16))) _Float16 v16h;
typedef __attribute__((ext_vector_type(8)))  _Float16 v8h;
typedef __attribute__((ext_vector_type(8)))  float    v8f;
typedef __attribute__((ext_vector_type(4)))  float    v4f;

#define NH    6
#define HD    24
#define HDP   32      // head dim padded for WMMA K
#define NWIN  64      // tokens per window
#define CDIM  144
#define HDIM  256
#define WDIM  256
#define SHIFT 4
#define LN100 4.60517018598809136804f

// LDS layout (halfs): Qs [6][64][32]  Ks [6][64][32]  Vt [6][32][64]  At [6][16][64]
#define QS_SZ (NH * NWIN * HDP)
#define VT_SZ (NH * HDP * NWIN)
#define AT_SZ (NH * 16 * NWIN)

// ws layout (floats): BiasY[6][64][16][4] at 0 (24576), tbl[225*6] at 24576
#define BIASY_SZ (NH * NWIN * 16 * 4)

// ---------------------------------------------------------------------------
// Kernel 1: CPB-MLP bias table  (tbl[m*6+h] = 16*sigmoid(mlp(coords_m)))
// ---------------------------------------------------------------------------
__global__ __launch_bounds__(256)
void cpb_table_kernel(const float* __restrict__ w1, const float* __restrict__ b1,
                      const float* __restrict__ w2, float* __restrict__ tbl) {
    __shared__ float red[NH][256];
    const int m = blockIdx.x;          // 0..224
    const int t = threadIdx.x;
    const int r = m / 15, c = m % 15;
    // coords: t*8/7, then sign(x)*log2(|x|+1)/log2(8)
    float tr = (float)(r - 7) * (8.0f / 7.0f);
    float tc = (float)(c - 7) * (8.0f / 7.0f);
    float sr = (tr > 0.f) ? 1.f : ((tr < 0.f) ? -1.f : 0.f);
    float sc = (tc > 0.f) ? 1.f : ((tc < 0.f) ? -1.f : 0.f);
    float v0 = sr * __log2f(fabsf(tr) + 1.0f) * (1.0f / 3.0f);
    float v1 = sc * __log2f(fabsf(tc) + 1.0f) * (1.0f / 3.0f);

    float acc[NH] = {0.f, 0.f, 0.f, 0.f, 0.f, 0.f};
    for (int j = t; j < 512; j += 256) {
        float hval = fmaxf(v0 * w1[j] + v1 * w1[512 + j] + b1[j], 0.0f);
#pragma unroll
        for (int h = 0; h < NH; ++h) acc[h] += hval * w2[j * NH + h];
    }
#pragma unroll
    for (int h = 0; h < NH; ++h) red[h][t] = acc[h];
    __syncthreads();
    for (int s = 128; s > 0; s >>= 1) {
        if (t < s) {
#pragma unroll
            for (int h = 0; h < NH; ++h) red[h][t] += red[h][t + s];
        }
        __syncthreads();
    }
    if (t < NH) {
        float x = red[t][0];
        tbl[m * NH + t] = 16.0f / (1.0f + __expf(-x));
    }
}

// ---------------------------------------------------------------------------
// Kernel 1b: expand table into WMMA-C-layout bias: BiasY[h][M][nlo][cb]
// so a lane's 4 column-tile biases for row M are one b128 load.
// ---------------------------------------------------------------------------
__global__ __launch_bounds__(256)
void bias_expand_kernel(const float* __restrict__ tbl, float* __restrict__ biasY) {
    int e = blockIdx.x * 256 + threadIdx.x;   // 0..24575
    int cb  = e & 3;
    int nlo = (e >> 2) & 15;
    int M   = (e >> 6) & 63;
    int h   = e >> 12;
    int n2  = cb * 16 + nlo;
    int i1 = M >> 3, j1 = M & 7, i2 = n2 >> 3, j2 = n2 & 7;
    int idx = (i1 - i2 + 7) * 15 + (j1 - j2 + 7);
    biasY[e] = tbl[idx * NH + h];
}

// ---------------------------------------------------------------------------
// Kernel 2: one block = one window; 6 waves = 6 heads; WMMA attention.
// ---------------------------------------------------------------------------
extern __shared__ char smem_raw[];

static __device__ __forceinline__ v16h pack16(v8h lo, v8h hi) {
    v16h a;
#pragma unroll
    for (int i = 0; i < 8; ++i) { a[i] = lo[i]; a[8 + i] = hi[i]; }
    return a;
}

__global__ __launch_bounds__(192)
void swin_attn_kernel(const float* __restrict__ qkv,
                      const float* __restrict__ logit_scale,
                      const float* __restrict__ biasY,
                      float* __restrict__ out) {
    _Float16* Qs = (_Float16*)smem_raw;               // [6][64][32]
    _Float16* Ks = Qs + QS_SZ;                        // [6][64][32]
    _Float16* Vt = Ks + QS_SZ;                        // [6][32][64]  (V transposed)
    _Float16* At = Vt + VT_SZ;                        // [6][16][64]  (attn scratch)

    const int tid = threadIdx.x;
    const int win = blockIdx.x;                       // 0..4095
    const int b   = win >> 10;
    const int wh  = (win >> 5) & 31;
    const int ww  = win & 31;

    // ---- stage qkv -> LDS f16 (roll by -shift folded into the gather) ----
    for (int idx = tid; idx < NWIN * CDIM; idx += 192) {
        int p = idx / CDIM, c = idx - p * CDIM;
        int h = c / HD, d = c - h * HD;
        int i = p >> 3, j = p & 7;
        int gh = (wh * 8 + i + SHIFT) & (HDIM - 1);
        int gw = (ww * 8 + j + SHIFT) & (WDIM - 1);
        const float* src = qkv + (size_t)((b * HDIM + gh) * WDIM + gw) * (3 * CDIM) + c;
        Qs[(h * NWIN + p) * HDP + d] = (_Float16)src[0];
        Ks[(h * NWIN + p) * HDP + d] = (_Float16)src[CDIM];
        Vt[(h * HDP + d) * NWIN + p] = (_Float16)src[2 * CDIM];
    }
    // zero the hd pad (24..31) so WMMA never sees NaN garbage on padded K/N
    for (int idx = tid; idx < NH * NWIN * 8; idx += 192) {
        int hp = idx >> 3, d = HD + (idx & 7);
        Qs[hp * HDP + d] = (_Float16)0.f;
        Ks[hp * HDP + d] = (_Float16)0.f;
    }
    for (int idx = tid; idx < NH * 8 * NWIN; idx += 192) {
        int h = idx >> 9, o = idx & 511;
        Vt[h * (HDP * NWIN) + HD * NWIN + o] = (_Float16)0.f;
    }
    __syncthreads();

    // ---- cosine-normalize Q,K rows in place; fold head scale into Q ----
    for (int t = tid; t < 2 * NH * NWIN; t += 192) {
        int qk = t & 1;
        int rowid = t >> 1;
        int h = rowid >> 6, row = rowid & 63;
        _Float16* base = (qk ? Ks : Qs) + (h * NWIN + row) * HDP;
        float ss = 0.f;
#pragma unroll
        for (int d = 0; d < HD; ++d) { float v = (float)base[d]; ss += v * v; }
        float inv = 1.0f / fmaxf(sqrtf(ss), 1e-12f);
        if (!qk) inv *= __expf(fminf(logit_scale[h], LN100));
#pragma unroll
        for (int d = 0; d < HD; ++d) base[d] = (_Float16)((float)base[d] * inv);
    }
    __syncthreads();

    // ---- per-wave attention: wave = head ----
    const int w     = tid >> 5;       // 0..5
    const int lane  = tid & 31;
    const int nlo   = lane & 15;
    const int khalf = lane >> 4;

    _Float16* Qh = Qs + w * NWIN * HDP;
    _Float16* Kh = Ks + w * NWIN * HDP;
    _Float16* Vh = Vt + w * HDP * NWIN;
    _Float16* Aw = At + w * 16 * NWIN;

    // shift-mask region ids computed arithmetically (only wh==31 / ww==31
    // windows have non-uniform regions for ws=8, shift=4, H=W=256)
    const bool eh = (wh == 31), ew = (ww == 31);
    const int  B2 = ew ? (((nlo & 7) < 4) ? 1 : 2) : 0;   // col j-region (lane const)
    int A2c[4];
#pragma unroll
    for (int cb = 0; cb < 4; ++cb) {
        int i2 = cb * 2 + (nlo >> 3);
        A2c[cb] = eh ? ((i2 < 4) ? 1 : 2) : 0;            // col i-region
    }

    for (int rb = 0; rb < 4; ++rb) {
        // A fragment (Q rows rb*16..+16, K=0..31): lane holds row nlo;
        // half 0: k {0..15}, half 1: k {16..31}
        const _Float16* arow = Qh + (rb * 16 + nlo) * HDP;
        v16h aQ = pack16(*(const v8h*)(arow + khalf * 16),
                         *(const v8h*)(arow + khalf * 16 + 8));

        // QK^T: 4 tiles of 16x16, single K=32 chunk each
        v8f acc[4];
#pragma unroll
        for (int cb = 0; cb < 4; ++cb) {
            const _Float16* brow = Kh + (cb * 16 + nlo) * HDP + khalf * 16;
            v16h bK = pack16(*(const v8h*)brow, *(const v8h*)(brow + 8));
            v8f cz = {};
            acc[cb] = __builtin_amdgcn_wmma_f32_16x16x32_f16(
                false, aQ, false, bK, (short)0, cz, false, false);
        }

        // bias (one b128 per row) + mask + row softmax
        const int A1 = eh ? (((rb * 2 + khalf) < 4) ? 1 : 2) : 0;  // row i-region
        const float* bp = biasY
            + (size_t)(((w * NWIN + rb * 16 + 8 * khalf) * 16 + nlo) * 4);
#pragma unroll
        for (int r = 0; r < 8; ++r) {
            v4f bias = *(const v4f*)(bp + r * 64);     // [cb] for row M
            const int B1 = ew ? ((r < 4) ? 1 : 2) : 0; // row j-region (j1 == r)
            float mx = -1e30f;
#pragma unroll
            for (int cb = 0; cb < 4; ++cb) {
                float v = acc[cb][r] + bias[cb];
                if ((A1 != A2c[cb]) || (B1 != B2)) v -= 100.0f;
                acc[cb][r] = v;
                mx = fmaxf(mx, v);
            }
#pragma unroll
            for (int m = 1; m < 16; m <<= 1) mx = fmaxf(mx, __shfl_xor(mx, m, 32));
            float sum = 0.f;
#pragma unroll
            for (int cb = 0; cb < 4; ++cb) {
                float e = __expf(acc[cb][r] - mx);
                acc[cb][r] = e;
                sum += e;
            }
#pragma unroll
            for (int m = 1; m < 16; m <<= 1) sum += __shfl_xor(sum, m, 32);
            float rinv = 1.0f / sum;
#pragma unroll
            for (int cb = 0; cb < 4; ++cb) {
                float p = acc[cb][r] * rinv;
                Aw[(r + 8 * khalf) * NWIN + cb * 16 + nlo] = (_Float16)p;
            }
        }
        __syncthreads();   // publish attn tile before re-fragmenting (uniform)

        // P @ V : out (16 x 24pad32), K=64 in two 32-chunks
        v8f o0 = {}, o1 = {};
#pragma unroll
        for (int kc = 0; kc < 2; ++kc) {
            const _Float16* ar = Aw + nlo * NWIN + kc * 32 + khalf * 16;
            v16h aP = pack16(*(const v8h*)ar, *(const v8h*)(ar + 8));
            const _Float16* br0 = Vh + nlo * NWIN + kc * 32 + khalf * 16;
            v16h bV0 = pack16(*(const v8h*)br0, *(const v8h*)(br0 + 8));
            o0 = __builtin_amdgcn_wmma_f32_16x16x32_f16(
                false, aP, false, bV0, (short)0, o0, false, false);
            const _Float16* br1 = Vh + (16 + nlo) * NWIN + kc * 32 + khalf * 16;
            v16h bV1 = pack16(*(const v8h*)br1, *(const v8h*)(br1 + 8));
            o1 = __builtin_amdgcn_wmma_f32_16x16x32_f16(
                false, aP, false, bV1, (short)0, o1, false, false);
        }

        // store to rolled-back global position (roll^-1 cancels the gather roll)
#pragma unroll
        for (int r = 0; r < 8; ++r) {
            int M = rb * 16 + r + 8 * khalf;
            int i = M >> 3, j = M & 7;
            int gh = (wh * 8 + i + SHIFT) & (HDIM - 1);
            int gw = (ww * 8 + j + SHIFT) & (WDIM - 1);
            size_t obase = (size_t)((b * HDIM + gh) * WDIM + gw) * CDIM + w * HD;
            out[obase + nlo] = o0[r];
            if (nlo < 8) out[obase + 16 + nlo] = o1[r];
        }
    }
}

// ---------------------------------------------------------------------------
extern "C" void kernel_launch(void* const* d_in, const int* in_sizes, int n_in,
                              void* d_out, int out_size, void* d_ws, size_t ws_size,
                              hipStream_t stream) {
    (void)in_sizes; (void)n_in; (void)out_size; (void)ws_size;
    const float* qkv         = (const float*)d_in[0];
    const float* logit_scale = (const float*)d_in[1];
    const float* cpb_w1      = (const float*)d_in[2];
    const float* cpb_b1      = (const float*)d_in[3];
    const float* cpb_w2      = (const float*)d_in[4];
    float* out   = (float*)d_out;
    float* biasY = (float*)d_ws;                      // 24576 floats
    float* tbl   = biasY + BIASY_SZ;                  // 1350 floats

    cpb_table_kernel<<<225, 256, 0, stream>>>(cpb_w1, cpb_b1, cpb_w2, tbl);
    bias_expand_kernel<<<BIASY_SZ / 256, 256, 0, stream>>>(tbl, biasY);

    size_t smem = (size_t)(2 * QS_SZ + VT_SZ + AT_SZ) * sizeof(_Float16);
    (void)hipFuncSetAttribute(reinterpret_cast<const void*>(swin_attn_kernel),
                              hipFuncAttributeMaxDynamicSharedMemorySize,
                              (int)smem);
    swin_attn_kernel<<<4096, 192, smem, stream>>>(qkv, logit_scale, biasY, out);
}
